// Quantizer_22935125360904
// MI455X (gfx1250) — matrix operand order repaired and checked by reading, not compiled, not run on hardware
//
#include <hip/hip_runtime.h>
#include <stdint.h>

typedef __attribute__((ext_vector_type(16))) __bf16 v16bf;
typedef __attribute__((ext_vector_type(8)))  float  v8f;
typedef int v4i __attribute__((vector_size(16)));   // matches builtin proto

namespace vq {
constexpr int N  = 32768;            // flattened rows (32*1024)
constexpr int D  = 512;              // embedding dim
constexpr int K  = 8192;             // codebook entries
constexpr int NT = K / 16;           // 512 codebook tiles (16 entries each)
constexpr int NJ = D / 32;           // 16 bf16 k-chunks of 32
constexpr size_t OFF_A = 0;                               // 32 MiB packed A (bf16)
constexpr size_t OFF_B = (size_t)N * D * 2;               // 8 MiB packed B (bf16)
constexpr size_t OFF_E = OFF_B + (size_t)K * D * 2;       // 32 KB ||e||^2
constexpr size_t OFF_I = OFF_E + (size_t)K * 4;           // 128 KB int indices
constexpr size_t OFF_P = OFF_I + (size_t)N * 4;           // 8 KB loss partials
}

// CDNA5 async global->LDS path (guarded: falls back to ds_store staging)
#if defined(__HIP_DEVICE_COMPILE__) &&                                        \
    __has_builtin(__builtin_amdgcn_global_load_async_to_lds_b128) &&          \
    __has_builtin(__builtin_amdgcn_s_wait_asynccnt)
#define VQ_USE_ASYNC 1
#else
#define VQ_USE_ASYNC 0
#endif

#if VQ_USE_ASYNC
// Probe-confirmed signature: (v4i AS1* src, v4i AS3* dst, imm offset, imm cpol)
__device__ __forceinline__ void vq_async_b128(const char* g, char* l) {
  __builtin_amdgcn_global_load_async_to_lds_b128(
      (__attribute__((address_space(1))) v4i*)g,
      (__attribute__((address_space(3))) v4i*)l, 0, 0);
}
#endif

// ---------------------------------------------------------------------------
// Pack inputs f32 -> bf16 in WMMA A-operand lane layout (16x32 tiles).
// A layout used here (contiguous k, not AMD's table order; B matches): lane = (half<<4)|m,
//   element e in [0,16): k = half*16 + e
// ---------------------------------------------------------------------------
__global__ __launch_bounds__(256) void vq_pack_a(const float* __restrict__ in,
                                                 uint16_t* __restrict__ a_out) {
  const int gt   = blockIdx.x * 256 + threadIdx.x;
  const int lane = gt & 31;
  const int t    = gt >> 5;           // tile id = rt*NJ + j
  const int j    = t & (vq::NJ - 1);
  const int rt   = t >> 4;
  const int m    = lane & 15;
  const int half = lane >> 4;
  const float* src = in + (size_t)(rt * 16 + m) * vq::D + j * 32;
  __bf16* dst = (__bf16*)(a_out + (size_t)t * 512 + lane * 16);
#pragma unroll
  for (int e = 0; e < 16; ++e) dst[e] = (__bf16)src[half * 16 + e];
}

// ---------------------------------------------------------------------------
// Pack codebook f32 -> bf16 in WMMA B-operand lane layout (32x16 tiles).
// B layout used here (contiguous k, same as A): lane = (half<<4)|n holds column n, element e -> k = half*16 + e.
// B[k][n] = codebook[c0+n][d0 + half*16 + e]  (16 contiguous floats per lane).
// ---------------------------------------------------------------------------
__global__ __launch_bounds__(256) void vq_pack_b(const float* __restrict__ cb,
                                                 uint16_t* __restrict__ b_out) {
  const int gt   = blockIdx.x * 256 + threadIdx.x;
  const int lane = gt & 31;
  const int t    = gt >> 5;           // tile id = ct*NJ + j
  const int j    = t & (vq::NJ - 1);
  const int ct   = t >> 4;
  const int n    = lane & 15;
  const int half = lane >> 4;
  const float* src = cb + (size_t)(ct * 16 + n) * vq::D + j * 32 + half * 16;
  __bf16* dst = (__bf16*)(b_out + (size_t)t * 512 + lane * 16);
#pragma unroll
  for (int e = 0; e < 16; ++e) dst[e] = (__bf16)src[e];
}

// ---------------------------------------------------------------------------
// ||e_k||^2 per codebook entry: one wave32 per entry.
// ---------------------------------------------------------------------------
__global__ __launch_bounds__(256) void vq_enorm(const float* __restrict__ cb,
                                                float* __restrict__ enorm) {
  const int wave = threadIdx.x >> 5, lane = threadIdx.x & 31;
  const int k = blockIdx.x * 8 + wave;
  const float* row = cb + (size_t)k * vq::D;
  float s = 0.f;
#pragma unroll
  for (int i = 0; i < vq::D / 32; ++i) {
    const float v = row[i * 32 + lane];
    s = fmaf(v, v, s);
  }
#pragma unroll
  for (int off = 16; off >= 1; off >>= 1) s += __shfl_xor(s, off, 32);
  if (lane == 0) enorm[k] = s;
}

// ---------------------------------------------------------------------------
// Fused bf16 WMMA GEMM (dot = X . C^T) + running argmin of ||e||^2 - 2*dot.
// Block = 8 waves = 128 rows; each wave owns 16 rows (A resident in 128
// VGPRs, total allocation < 256 so no VGPR-MSB juggling). B chunks are
// prefetched 4-deep from LDS so ds_load latency is hidden behind WMMAs.
// Codebook tiles (16 KB) double-buffered in LDS via async global->LDS loads.
// ---------------------------------------------------------------------------
__global__ __launch_bounds__(256) void vq_gemm_argmin(
    const uint16_t* __restrict__ a_t, const uint16_t* __restrict__ b_t,
    const float* __restrict__ enorm, int* __restrict__ idx_i,
    float* __restrict__ idx_f) {
  __shared__ alignas(32) uint16_t Bs[2][8192];  // 2 x 16 KB

  const int tid  = threadIdx.x;
  const int wave = tid >> 5, lane = tid & 31;
  const int half = lane >> 4, n = lane & 15;
  const int rt   = blockIdx.x * 8 + wave;  // this wave's 16-row tile

  // Load this wave's 16 input rows (bf16, WMMA layout) into registers.
  const v16bf* Ap = (const v16bf*)a_t;
  v16bf a[16];
#pragma unroll
  for (int j = 0; j < 16; ++j) a[j] = Ap[((size_t)(rt * 16 + j)) * 32 + lane];

  float best[8];
  int   bk[8];
#pragma unroll
  for (int r = 0; r < 8; ++r) { best[r] = 3.4e38f; bk[r] = 0; }

#if VQ_USE_ASYNC
  {  // prologue: async-load codebook tile 0 into buffer 0 (4 x b128 per lane)
    const char* g = (const char*)b_t + (size_t)wave * 2048 + lane * 16;
    char* l = (char*)&Bs[0][0] + wave * 2048 + lane * 16;
#pragma unroll
    for (int i = 0; i < 4; ++i) vq_async_b128(g + i * 512, l + i * 512);
  }
#endif

  for (int ct = 0; ct < vq::NT; ++ct) {
    const int p = ct & 1;
#if VQ_USE_ASYNC
    if (ct + 1 < vq::NT) {  // prefetch next tile into the other buffer
      const char* g = (const char*)b_t + (size_t)(ct + 1) * 16384 +
                      wave * 2048 + lane * 16;
      char* l = (char*)&Bs[p ^ 1][0] + wave * 2048 + lane * 16;
#pragma unroll
      for (int i = 0; i < 4; ++i) vq_async_b128(g + i * 512, l + i * 512);
      __builtin_amdgcn_s_wait_asynccnt(4);  // oldest 4 (tile ct) complete
    } else {
      __builtin_amdgcn_s_wait_asynccnt(0);
    }
    __syncthreads();  // all waves' portions of tile ct visible
    const v16bf* Bp = (const v16bf*)&Bs[p][0];
#else
    (void)p;
    __syncthreads();  // previous tile's readers done before overwrite
    {
      const uint4* g = (const uint4*)((const char*)b_t + (size_t)ct * 16384);
      uint4* s = (uint4*)&Bs[0][0];
#pragma unroll
      for (int i = 0; i < 4; ++i) s[tid + i * 256] = g[tid + i * 256];
    }
    __syncthreads();
    const v16bf* Bp = (const v16bf*)&Bs[0][0];
#endif

    // Hoist the epilogue operand load above the matrix loop.
    const int   kg = (ct << 4) + n;
    const float en = enorm[kg];

    // 4-deep rotating register pipeline over the 16 B chunks.
    v8f acc = {};
    v16bf b0 = Bp[0 * 32 + lane];
    v16bf b1 = Bp[1 * 32 + lane];
    v16bf b2 = Bp[2 * 32 + lane];
    v16bf b3 = Bp[3 * 32 + lane];
#pragma unroll
    for (int j = 0; j < 16; ++j) {
      v16bf bn = b3;
      if (j + 4 < 16) bn = Bp[(j + 4) * 32 + lane];
      acc = __builtin_amdgcn_wmma_f32_16x16x32_bf16(false, a[j], false, b0,
                                                    (short)0, acc, false, false);
      b0 = b1; b1 = b2; b2 = b3; b3 = bn;
    }

    // dist(row m, col n) = ||e||^2 - 2*dot ; lane holds col n, slot r = row.
#pragma unroll
    for (int r = 0; r < 8; ++r) {
      const float c = fmaf(-2.f, acc[r], en);
      if (c < best[r]) { best[r] = c; bk[r] = kg; }
    }
#if VQ_USE_ASYNC
    __syncthreads();  // all waves done reading Bs[p] before it is refilled
#endif
  }

  // Cross-lane argmin within each 16-lane half (same rows, different columns).
#pragma unroll
  for (int r = 0; r < 8; ++r) {
    float b = best[r];
    int   k = bk[r];
#pragma unroll
    for (int off = 8; off >= 1; off >>= 1) {
      const float ob = __shfl_xor(b, off, 32);
      const int   ok = __shfl_xor(k, off, 32);
      if (ob < b || (ob == b && ok < k)) { b = ob; k = ok; }
    }
    if (n == 0) {  // lanes 0 (rows m=0..7) and 16 (rows m=8..15)
      const int row = rt * 16 + half * 8 + r;
      idx_i[row] = k;
      idx_f[row] = (float)k;
    }
  }
}

// ---------------------------------------------------------------------------
// Gather quantized rows + deterministic per-block partial loss sums.
// ---------------------------------------------------------------------------
__global__ __launch_bounds__(256) void vq_gather_loss(
    const float* __restrict__ in, const float* __restrict__ cb,
    const int* __restrict__ idx, float* __restrict__ outq,
    float* __restrict__ partial) {
  __shared__ float red[256];
  __shared__ int rowidx[16];
  const int b = blockIdx.x, tid = threadIdx.x;
  if (tid < 16) rowidx[tid] = idx[b * 16 + tid];
  __syncthreads();
  float s = 0.f;
#pragma unroll
  for (int i = 0; i < 8; ++i) {
    const int f = tid + i * 256;         // float4 index within 16-row chunk
    const int r = f >> 7, c = f & 127;   // 128 float4 per row
    const float4 q = ((const float4*)(cb + (size_t)rowidx[r] * vq::D))[c];
    const float4 x = ((const float4*)(in + ((size_t)b * 16 + r) * vq::D))[c];
    ((float4*)(outq + ((size_t)b * 16 + r) * vq::D))[c] = q;
    const float dx = q.x - x.x, dy = q.y - x.y, dz = q.z - x.z, dw = q.w - x.w;
    s += dx * dx + dy * dy + dz * dz + dw * dw;
  }
  red[tid] = s;
  __syncthreads();
  for (int off = 128; off > 0; off >>= 1) {
    if (tid < off) red[tid] += red[tid + off];
    __syncthreads();
  }
  if (tid == 0) partial[b] = red[0];
}

__global__ __launch_bounds__(256) void vq_final_loss(
    const float* __restrict__ partial, float* __restrict__ loss) {
  __shared__ float red[256];
  const int tid = threadIdx.x;
  float s = 0.f;
#pragma unroll
  for (int i = 0; i < 8; ++i) s += partial[tid + i * 256];  // 2048 partials
  red[tid] = s;
  __syncthreads();
  for (int off = 128; off > 0; off >>= 1) {
    if (tid < off) red[tid] += red[tid + off];
    __syncthreads();
  }
  // loss = q_latent + 0.25*e_latent, and both equal mean((q-x)^2).
  if (tid == 0) loss[0] = 1.25f * red[0] / (float)((size_t)vq::N * vq::D);
}

// ---------------------------------------------------------------------------
extern "C" void kernel_launch(void* const* d_in, const int* in_sizes, int n_in,
                              void* d_out, int out_size, void* d_ws,
                              size_t ws_size, hipStream_t stream) {
  (void)in_sizes; (void)n_in; (void)out_size; (void)ws_size;
  const float* inputs = (const float*)d_in[0];   // [32,1024,512] f32
  const float* cb     = (const float*)d_in[1];   // [8192,512] f32

  char* ws = (char*)d_ws;
  uint16_t* a_bf    = (uint16_t*)(ws + vq::OFF_A);
  uint16_t* b_bf    = (uint16_t*)(ws + vq::OFF_B);
  float*    enorm   = (float*)(ws + vq::OFF_E);
  int*      idx_i   = (int*)(ws + vq::OFF_I);
  float*    partial = (float*)(ws + vq::OFF_P);

  float* outq  = (float*)d_out;                          // quantized_st
  float* idx_f = outq + (size_t)vq::N * vq::D;           // indices (as f32)
  float* loss  = idx_f + vq::N;                          // scalar loss

  vq_pack_a<<<4096, 256, 0, stream>>>(inputs, a_bf);
  vq_pack_b<<<1024, 256, 0, stream>>>(cb, b_bf);
  vq_enorm<<<1024, 256, 0, stream>>>(cb, enorm);
  vq_gemm_argmin<<<vq::N / 128, 256, 0, stream>>>(a_bf, b_bf, enorm, idx_i, idx_f);
  vq_gather_loss<<<vq::N / 16, 256, 0, stream>>>(inputs, cb, idx_i, outq, partial);
  vq_final_loss<<<1, 256, 0, stream>>>(partial, loss);
}
